// PointNetFeaturePropagation_3650722202243
// MI455X (gfx1250) — matrix operand order repaired and checked by reading, not compiled, not run on hardware
//
#include <hip/hip_runtime.h>
#include <hip/hip_bf16.h>

// ---------- problem constants ----------
constexpr int BB  = 8;
constexpr int NN  = 8192;
constexpr int MM  = 2048;
constexpr int C1  = 128;
constexpr int C2  = 256;
constexpr int RR  = BB * NN;        // 65536 rows
constexpr int K0  = C1 + C2;        // 384
constexpr int O0  = 256;
constexpr int O1  = 128;
constexpr int PBLK = 256;                 // stat partial blocks
constexpr int ROWS_PER_STAT = RR / PBLK;  // 256
constexpr float EPS_W  = 1e-8f;
constexpr float EPS_BN = 1e-5f;

typedef __attribute__((ext_vector_type(16))) _Float16 v16h;
typedef __attribute__((ext_vector_type(8)))  _Float16 v8h;
typedef __attribute__((ext_vector_type(8)))  float    v8f;

// ---------- f32 -> f16 convert ----------
__global__ __launch_bounds__(256) void cvt_f16_kernel(const float* __restrict__ src,
                                                      _Float16* __restrict__ dst, int n) {
  int i = blockIdx.x * 256 + threadIdx.x;
  if (i < n) dst[i] = (_Float16)src[i];
}

// ---------- 3-NN + inverse-distance weights ----------
__global__ __launch_bounds__(256) void top3_kernel(const float* __restrict__ xyz1,
                                                   const float* __restrict__ xyz2,
                                                   int* __restrict__ idx,
                                                   float* __restrict__ wgt) {
  __shared__ float s2[MM * 3];                 // 24 KB: whole xyz2[b] tile
  const int p = blockIdx.x * 256 + threadIdx.x;  // global point (8192/256 blocks per batch)
  const int b = p / NN;
  for (int i = threadIdx.x; i < MM * 3; i += 256)
    s2[i] = xyz2[(size_t)b * MM * 3 + i];
  __syncthreads();

  const float x = xyz1[(size_t)p * 3 + 0];
  const float y = xyz1[(size_t)p * 3 + 1];
  const float z = xyz1[(size_t)p * 3 + 2];

  float d0 = 1e30f, d1 = 1e30f, d2 = 1e30f;
  int   i0 = 0,     i1 = 0,     i2 = 0;
  for (int m = 0; m < MM; ++m) {
    const float dx = s2[m * 3 + 0] - x;   // uniform m -> LDS broadcast, no bank conflicts
    const float dy = s2[m * 3 + 1] - y;
    const float dz = s2[m * 3 + 2] - z;
    const float d  = dx * dx + dy * dy + dz * dz;
    if (d < d0)      { d2 = d1; i2 = i1; d1 = d0; i1 = i0; d0 = d; i0 = m; }
    else if (d < d1) { d2 = d1; i2 = i1; d1 = d;  i1 = m; }
    else if (d < d2) { d2 = d;  i2 = m; }
  }
  const float r0 = 1.0f / (d0 + EPS_W);
  const float r1 = 1.0f / (d1 + EPS_W);
  const float r2 = 1.0f / (d2 + EPS_W);
  const float inv = 1.0f / (r0 + r1 + r2);
  idx[(size_t)p * 3 + 0] = i0;  wgt[(size_t)p * 3 + 0] = r0 * inv;
  idx[(size_t)p * 3 + 1] = i1;  wgt[(size_t)p * 3 + 1] = r1 * inv;
  idx[(size_t)p * 3 + 2] = i2;  wgt[(size_t)p * 3 + 2] = r2 * inv;
}

// ---------- interpolate + concat -> x[R, 384] f16 ----------
__global__ __launch_bounds__(384) void build_x_kernel(const float* __restrict__ points1,
                                                      const float* __restrict__ points2,
                                                      const int* __restrict__ idx,
                                                      const float* __restrict__ wgt,
                                                      _Float16* __restrict__ Xh) {
  const int r = blockIdx.x;            // one point per block
  const int b = r / NN;
  const int t = threadIdx.x;
  if (t < C1) {
    Xh[(size_t)r * K0 + t] = (_Float16)points1[(size_t)r * C1 + t];
  } else {
    const int c = t - C1;
    const int i0 = idx[(size_t)r * 3 + 0];
    const int i1 = idx[(size_t)r * 3 + 1];
    const int i2 = idx[(size_t)r * 3 + 2];
    const float w0 = wgt[(size_t)r * 3 + 0];
    const float w1 = wgt[(size_t)r * 3 + 1];
    const float w2 = wgt[(size_t)r * 3 + 2];
    const float* P2 = points2 + (size_t)b * MM * C2;
    const float v = w0 * P2[(size_t)i0 * C2 + c]
                  + w1 * P2[(size_t)i1 * C2 + c]
                  + w2 * P2[(size_t)i2 * C2 + c];
    Xh[(size_t)r * K0 + t] = (_Float16)v;
  }
}

// ---------- WMMA GEMM: Y[R,O] = X[R,K](f16) @ W[O,K]^T(f16) + bias ----------
// 8 waves/block; wave w owns 16 rows. Weights staged through LDS in 64-channel
// slabs (padded stride -> conflict-free ds_load_b128 B-fragment reads).
template <int KDIM, int ODIM, bool HALF_OUT>
__global__ __launch_bounds__(256) void gemm_kernel(const _Float16* __restrict__ X,
                                                   const _Float16* __restrict__ Wh,
                                                   const float* __restrict__ bias,
                                                   void* __restrict__ out) {
  constexpr int OCH  = 64;           // output channels per LDS slab
  constexpr int SROW = KDIM + 8;     // padded LDS row stride (halves): 784B = 196 dw -> +4 banks/lane
  constexpr int SEGS = KDIM / 8;     // uint4 segments per weight row
  __shared__ _Float16 sW[OCH * SROW];

  const int wave = threadIdx.x >> 5;
  const int lane = threadIdx.x & 31;
  const int m    = lane & 15;   // row within A-tile / col within B-tile
  const int h    = lane >> 4;   // lane half
  const int r0   = (blockIdx.x * 8 + wave) * 16;

  for (int oc = 0; oc < ODIM; oc += OCH) {
    __syncthreads();   // previous slab fully consumed
    // cooperative stage: W rows [oc, oc+64) x KDIM  ->  LDS (padded rows)
    for (int i = threadIdx.x; i < OCH * SEGS; i += 256) {
      const int row = i / SEGS;
      const int seg = i - row * SEGS;
      ((uint4*)(sW + row * SROW))[seg] =
          ((const uint4*)(Wh + (size_t)(oc + row) * KDIM))[seg];
    }
    __syncthreads();

    for (int ot = 0; ot < OCH; ot += 16) {
      const float cz = bias[oc + ot + m];
      v8f c;
#pragma unroll
      for (int v = 0; v < 8; ++v) c[v] = cz;

#pragma unroll
      for (int kk = 0; kk < KDIM; kk += 32) {
        // A 16x32 f16: elems 0..7 -> K = kk + h*8 + e ; elems 8..15 -> K = kk + 16 + h*8 + e
        const _Float16* pa = X + (size_t)(r0 + m) * KDIM + kk + h * 8;
        v8h alo = *(const v8h*)(pa);
        v8h ahi = *(const v8h*)(pa + 16);
        v16h a = __builtin_shufflevector(alo, ahi, 0, 1, 2, 3, 4, 5, 6, 7,
                                         8, 9, 10, 11, 12, 13, 14, 15);
        // B 32x16 f16 from LDS: elems 0..7 -> K = kk + h*8 + e ; elems 8..15 -> K = kk + 16 + h*8 + e, col = oc + ot + m
        const _Float16* pb = sW + (ot + m) * SROW + kk + h * 8;
        v8h blo = *(const v8h*)(pb);
        v8h bhi = *(const v8h*)(pb + 16);
        v16h bfr = __builtin_shufflevector(blo, bhi, 0, 1, 2, 3, 4, 5, 6, 7,
                                           8, 9, 10, 11, 12, 13, 14, 15);

        c = __builtin_amdgcn_wmma_f32_16x16x32_f16(false, a, false, bfr,
                                                   (short)0, c, false, false);
      }
      // C/D: elem v -> row r0 + v + 8*h, col oc + ot + m
      if (HALF_OUT) {
        _Float16* po = (_Float16*)out;
#pragma unroll
        for (int v = 0; v < 8; ++v)
          po[(size_t)(r0 + v + 8 * h) * ODIM + oc + ot + m] = (_Float16)c[v];
      } else {
        float* po = (float*)out;
#pragma unroll
        for (int v = 0; v < 8; ++v)
          po[(size_t)(r0 + v + 8 * h) * ODIM + oc + ot + m] = c[v];
      }
    }
  }
}

// ---------- BN stats: per-block partial sums (deterministic 2-stage) ----------
template <typename T, int ODIM>
__global__ void stats_kernel(const T* __restrict__ y,
                             float* __restrict__ psum, float* __restrict__ psq) {
  const int t  = threadIdx.x;          // channel (blockDim == ODIM)
  const int r0 = blockIdx.x * ROWS_PER_STAT;
  float s = 0.0f, ss = 0.0f;
  for (int i = 0; i < ROWS_PER_STAT; ++i) {
    const float v = (float)y[(size_t)(r0 + i) * ODIM + t];
    s += v; ss += v * v;
  }
  psum[(size_t)blockIdx.x * ODIM + t] = s;
  psq [(size_t)blockIdx.x * ODIM + t] = ss;
}

template <int ODIM>
__global__ void finalize_kernel(const float* __restrict__ psum, const float* __restrict__ psq,
                                const float* __restrict__ gamma, const float* __restrict__ beta,
                                float* __restrict__ scale, float* __restrict__ shift) {
  const int t = threadIdx.x;
  float s = 0.0f, ss = 0.0f;
  for (int p = 0; p < PBLK; ++p) {   // fixed order -> deterministic
    s  += psum[(size_t)p * ODIM + t];
    ss += psq [(size_t)p * ODIM + t];
  }
  const float mean = s / (float)RR;
  const float var  = ss / (float)RR - mean * mean;
  const float inv  = rsqrtf(var + EPS_BN);
  const float sc   = gamma[t] * inv;
  scale[t] = sc;
  shift[t] = beta[t] - mean * sc;
}

// ---------- normalize + ReLU (elementwise, may be in place) ----------
template <typename TIn, typename TOut, int ODIM>
__global__ __launch_bounds__(256) void norm_kernel(const TIn* __restrict__ in,
                                                   TOut* __restrict__ out,
                                                   const float* __restrict__ scale,
                                                   const float* __restrict__ shift) {
  const size_t i = (size_t)blockIdx.x * 256 + threadIdx.x;
  const int c = (int)(i & (ODIM - 1));
  float v = (float)in[i] * scale[c] + shift[c];
  v = v > 0.0f ? v : 0.0f;
  out[i] = (TOut)v;
}

// ---------- host launcher ----------
extern "C" void kernel_launch(void* const* d_in, const int* in_sizes, int n_in,
                              void* d_out, int out_size, void* d_ws, size_t ws_size,
                              hipStream_t stream) {
  const float* xyz1    = (const float*)d_in[0];
  const float* xyz2    = (const float*)d_in[1];
  const float* points1 = (const float*)d_in[2];
  const float* points2 = (const float*)d_in[3];
  const float* W0      = (const float*)d_in[4];
  const float* b0      = (const float*)d_in[5];
  const float* gamma0  = (const float*)d_in[6];
  const float* beta0   = (const float*)d_in[7];
  const float* W1      = (const float*)d_in[8];
  const float* b1      = (const float*)d_in[9];
  const float* gamma1  = (const float*)d_in[10];
  const float* beta1   = (const float*)d_in[11];

  char* ws = (char*)d_ws;
  size_t off = 0;
  _Float16* W0h = (_Float16*)(ws + off); off += (size_t)O0 * K0 * 2;        // 196608
  _Float16* W1h = (_Float16*)(ws + off); off += (size_t)O1 * O0 * 2;        // 65536
  int*      idx = (int*)     (ws + off); off += (size_t)RR * 3 * 4;         // 786432
  float*    wgt = (float*)   (ws + off); off += (size_t)RR * 3 * 4;         // 786432
  _Float16* Xh  = (_Float16*)(ws + off); off += (size_t)RR * K0 * 2;        // 50331648
  _Float16* y0h = (_Float16*)(ws + off); off += (size_t)RR * O0 * 2;        // 33554432
  float*   psum = (float*)   (ws + off); off += (size_t)PBLK * O0 * 4;      // 262144
  float*   psq  = (float*)   (ws + off); off += (size_t)PBLK * O0 * 4;      // 262144
  float*  scale = (float*)   (ws + off); off += 256 * 4;
  float*  shift = (float*)   (ws + off); off += 256 * 4;

  float* out = (float*)d_out;   // [R, 128] f32

  // weights -> f16
  cvt_f16_kernel<<<(O0 * K0 + 255) / 256, 256, 0, stream>>>(W0, W0h, O0 * K0);
  cvt_f16_kernel<<<(O1 * O0 + 255) / 256, 256, 0, stream>>>(W1, W1h, O1 * O0);

  // 3-NN + weights
  top3_kernel<<<RR / 256, 256, 0, stream>>>(xyz1, xyz2, idx, wgt);

  // interpolate + concat -> Xh[R,384] f16
  build_x_kernel<<<RR, 384, 0, stream>>>(points1, points2, idx, wgt, Xh);

  // layer 0: GEMM (WMMA, LDS-staged weights) -> y0h f16, then BN stats / norm / ReLU
  gemm_kernel<K0, O0, true><<<RR / 128, 256, 0, stream>>>(Xh, W0h, b0, (void*)y0h);
  stats_kernel<_Float16, O0><<<PBLK, O0, 0, stream>>>(y0h, psum, psq);
  finalize_kernel<O0><<<1, O0, 0, stream>>>(psum, psq, gamma0, beta0, scale, shift);
  norm_kernel<_Float16, _Float16, O0><<<(RR * O0) / 256, 256, 0, stream>>>(y0h, y0h, scale, shift);

  // layer 1: GEMM (WMMA) -> d_out f32 (pre-BN), then stats / norm / ReLU in place
  gemm_kernel<O0, O1, false><<<RR / 128, 256, 0, stream>>>(y0h, W1h, b1, (void*)out);
  stats_kernel<float, O1><<<PBLK, O1, 0, stream>>>(out, psum, psq);
  finalize_kernel<O1><<<1, O1, 0, stream>>>(psum, psq, gamma1, beta1, scale, shift);
  norm_kernel<float, float, O1><<<(RR * O1) / 256, 256, 0, stream>>>(out, out, scale, shift);
}
